// CrossModalFusion_83580063580672
// MI455X (gfx1250) — matrix operand order repaired and checked by reading, not compiled, not run on hardware
//
#include <hip/hip_runtime.h>

typedef __attribute__((ext_vector_type(16))) _Float16 v16h;
typedef __attribute__((ext_vector_type(8)))  _Float16 v8h;
typedef __attribute__((ext_vector_type(8)))  float    v8f;

// ---------------- input leaf indices (JAX pytree: sorted dict keys, DFS) ----
enum {
  IN_IMG_GLOBAL = 0, IN_IMG_PATCH = 1,
  I2T_KW=2, I2T_KB=3, I2T_OW=4, I2T_OB=5, I2T_QW=6, I2T_QB=7, I2T_VW=8, I2T_VB=9,
  T2I_KW=10, T2I_KB=11, T2I_OW=12, T2I_OB=13, T2I_QW=14, T2I_QB=15, T2I_VW=16, T2I_VB=17,
  IG_ALPHA=18, IG_LNB=19, IG_LNG=20, IG_KW=21, IG_KB=22, IG_OW=23, IG_OB=24,
  IG_QW=25, IG_QB=26, IG_VW=27, IG_VB=28, IG_POS=29,
  GPROJ_W=30, GPROJ_B=31, PPROJ_W=32, PPROJ_B=33,
  IP_ALPHA=34, IP_LNB=35, IP_LNG=36, IP_KW=37, IP_KB=38, IP_OW=39, IP_OB=40,
  IP_QW=41, IP_QB=42, IP_VW=43, IP_VB=44, IP_POS=45,
  KIMG_W=46, KIMG_B=47, KTXT_W=48, KTXT_B=49,
  LNIMG_B=50, LNIMG_G=51, LNTXT_B=52, LNTXT_G=53,
  QIMG_W=54, QIMG_B=55, QTXT_W=56, QTXT_B=57,
  TXT_ALPHA=58, TXT_LNB=59, TXT_LNG=60, TXT_KW=61, TXT_KB=62, TXT_OW=63, TXT_OB=64,
  TXT_QW=65, TXT_QB=66, TXT_VW=67, TXT_VB=68, TXT_POS=69,
  TPROJ_W=70, TPROJ_B=71, VIMG_W=72, VIMG_B=73, VTXT_W=74, VTXT_B=75,
  IN_TXT=76
};

// ============================================================================
// Batched WMMA GEMM, f16 operands, f32 accumulate:  C = A*B^T (+bias) (+R)
// A row-major f16: Ab + m*lda + k
// B "row n" K-contiguous:  Bb + n*ldb + k   (weights (N,K) or pre-transposed V)
// cfmt: 0 -> f32 stores, 1 -> f16 stores.
// batch z = blockIdx.y; b=z/H, h=z%H; base offsets b*s?b + h*s?h (elements).
// Block = 8 waves sharing one 32-row A strip. The A tile is DOUBLE-BUFFERED
// in LDS via global_load_async_to_lds_b128 (ASYNCcnt): tile k+1's DMA overlaps
// the WMMA work on tile k; one workgroup barrier per K-step.
// All of M,N,K are multiples of 32 in this model.
// ============================================================================
__global__ __launch_bounds__(256)
void k_gemm(const _Float16* __restrict__ A, int lda, long sAb, long sAh,
            const _Float16* __restrict__ Bw, int ldb, long sBb, long sBh,
            const float* __restrict__ bias,
            const _Float16* __restrict__ Rres, int ldr, long sRb, long sRh,
            void* __restrict__ C, int ldc, long sCb, long sCh, int cfmt,
            int M, int N, int K, int H)
{
  __shared__ __align__(16) _Float16 sA[2][32 * 40]; // 2 bufs, 40-half row stride (bank-safe)
  const int t    = threadIdx.x;
  const int wave = t >> 5;
  const int lane = t & 31;
  const int Nt  = N >> 5;
  const int NtG = (Nt + 7) >> 3;
  const int bm  = (int)blockIdx.x / NtG;
  const int tn  = ((int)blockIdx.x % NtG) * 8 + wave;
  const bool act = tn < Nt;                       // wave-uniform
  const int z = blockIdx.y;
  const int b = z / H, h = z % H;
  const _Float16* Ab = A  + b * sAb + h * sAh;
  const _Float16* Bb = Bw + b * sBb + h * sBh;
  const int m0 = bm << 5, n0 = tn << 5;
  const int lr = lane & 15;
  const int lh = lane >> 4;

  // Async staging: threads 0..127 move 16B each (32 rows x 64B tile)
  const int arow = t >> 2;
  const int acol = (t & 3) * 8;                    // halves
  const unsigned bufBytes = 32 * 40 * 2;
  const unsigned ldsDst0 =
      (unsigned)(unsigned long long)(const void*)&sA[0][0] +
      (unsigned)(arow * 40 + acol) * 2;
  const _Float16* gsrc = Ab + (unsigned)((m0 + arow) * lda + acol);

  const _Float16* pB0 = Bb + (unsigned)((n0 + lr)      * ldb + lh * 16);
  const _Float16* pB1 = Bb + (unsigned)((n0 + 16 + lr) * ldb + lh * 16);

  v8f c00 = {}, c01 = {}, c10 = {}, c11 = {};

  // preload tile 0 into buffer 0
  if (t < 128) {
    asm volatile("global_load_async_to_lds_b128 %0, %1, off"
                 :: "v"(ldsDst0), "v"(gsrc) : "memory");
  }

  int buf = 0;
  for (int k0 = 0; k0 < K; k0 += 32, buf ^= 1) {
    asm volatile("s_wait_asynccnt 0x0" ::: "memory");
    __syncthreads();                      // publish tile k0; buf^1 readers done
    if (k0 + 32 < K && t < 128) {         // overlap next tile's DMA with compute
      asm volatile("global_load_async_to_lds_b128 %0, %1, off"
                   :: "v"(ldsDst0 + (unsigned)(buf ^ 1) * bufBytes),
                      "v"(gsrc + k0 + 32) : "memory");
    }
    if (act) {
      const _Float16* sb = &sA[buf][0];
      v8h a0lo = *(const v8h*)&sb[lr * 40 + lh * 16];
      v8h a0hi = *(const v8h*)&sb[lr * 40 + 8 + lh * 16];
      v8h a1lo = *(const v8h*)&sb[(lr + 16) * 40 + lh * 16];
      v8h a1hi = *(const v8h*)&sb[(lr + 16) * 40 + 8 + lh * 16];
      v16h a0 = __builtin_shufflevector(a0lo, a0hi, 0,1,2,3,4,5,6,7,8,9,10,11,12,13,14,15);
      v16h a1 = __builtin_shufflevector(a1lo, a1hi, 0,1,2,3,4,5,6,7,8,9,10,11,12,13,14,15);
      v16h b0 = *(const v16h*)(pB0 + k0);
      v16h b1 = *(const v16h*)(pB1 + k0);
      c00 = __builtin_amdgcn_wmma_f32_16x16x32_f16(false, a0, false, b0, (short)0, c00, false, false);
      c01 = __builtin_amdgcn_wmma_f32_16x16x32_f16(false, a0, false, b1, (short)0, c01, false, false);
      c10 = __builtin_amdgcn_wmma_f32_16x16x32_f16(false, a1, false, b0, (short)0, c10, false, false);
      c11 = __builtin_amdgcn_wmma_f32_16x16x32_f16(false, a1, false, b1, (short)0, c11, false, false);
    }
  }
  if (!act) return;

  // Epilogue: C/D layout (m = r + 8*lh, n = lr)
  const int cn0 = n0 + lr, cn1 = cn0 + 16;
  const float bi0 = bias ? bias[cn0] : 0.f;
  const float bi1 = bias ? bias[cn1] : 0.f;
  const _Float16* Rb = Rres ? (Rres + b * sRb + h * sRh) : nullptr;
  const long cbase = b * sCb + h * sCh;
#pragma unroll
  for (int r = 0; r < 8; ++r) {
    const int m  = m0 + r + lh * 8;
    const int m2 = m + 16;
    float v00 = c00[r] + bi0, v01 = c01[r] + bi1;
    float v10 = c10[r] + bi0, v11 = c11[r] + bi1;
    if (Rb) {
      v00 += (float)Rb[(unsigned)(m  * ldr + cn0)];
      v01 += (float)Rb[(unsigned)(m  * ldr + cn1)];
      v10 += (float)Rb[(unsigned)(m2 * ldr + cn0)];
      v11 += (float)Rb[(unsigned)(m2 * ldr + cn1)];
    }
    if (cfmt == 0) {
      float* Cb = (float*)C + cbase;
      Cb[(unsigned)(m  * ldc + cn0)] = v00;
      Cb[(unsigned)(m  * ldc + cn1)] = v01;
      Cb[(unsigned)(m2 * ldc + cn0)] = v10;
      Cb[(unsigned)(m2 * ldc + cn1)] = v11;
    } else {
      _Float16* Cb = (_Float16*)C + cbase;
      Cb[(unsigned)(m  * ldc + cn0)] = (_Float16)v00;
      Cb[(unsigned)(m  * ldc + cn1)] = (_Float16)v01;
      Cb[(unsigned)(m2 * ldc + cn0)] = (_Float16)v10;
      Cb[(unsigned)(m2 * ldc + cn1)] = (_Float16)v11;
    }
  }
}

// ---------------- head transpose: V (B,L,H*dh) -> VT (B,H,dh,L) -------------
__global__ __launch_bounds__(256)
void k_transpose(const _Float16* __restrict__ in, _Float16* __restrict__ out,
                 int L, int D, int dh, int H)
{
  __shared__ _Float16 tile[32][33];
  const int tx = threadIdx.x & 31;
  const int ty = threadIdx.x >> 5;                 // 0..7
  const int k0 = blockIdx.x * 32;
  const int n0 = blockIdx.y * 32;
  const int g  = blockIdx.z;                        // b*H + h
  const int b = g / H, h = g % H;
  const _Float16* ib = in + (long)b * L * D + h * dh;
  _Float16* ob = out + (long)g * dh * L;
#pragma unroll
  for (int r = 0; r < 4; ++r) {
    int kk = ty + 8 * r;
    tile[kk][tx] = ib[(unsigned)((k0 + kk) * D + n0 + tx)];
  }
  __syncthreads();
#pragma unroll
  for (int r = 0; r < 4; ++r) {
    int nn = ty + 8 * r;
    ob[(unsigned)((n0 + nn) * L + k0 + tx)] = tile[tx][nn];
  }
}

// ---------------- elementwise / row kernels ---------------------------------
__global__ void k_cvt(const float* __restrict__ in, _Float16* __restrict__ out, long n)
{
  long i = (long)blockIdx.x * blockDim.x + threadIdx.x;
  if (i < n) out[i] = (_Float16)in[i];
}

__global__ void k_add_pos(const float* __restrict__ in, const float* __restrict__ pos,
                          _Float16* __restrict__ out, long total, int LD)
{
  long i = (long)blockIdx.x * blockDim.x + threadIdx.x;
  if (i < total) out[i] = (_Float16)(in[i] + pos[i % LD]);
}

// row-wise softmax: reads f32 scores, writes f16 probabilities (A of P*V GEMM)
__global__ __launch_bounds__(256)
void k_softmax(const float* __restrict__ S, _Float16* __restrict__ Pm, int C, float scale)
{
  long row = blockIdx.x;
  const float* x = S + row * (long)C;
  _Float16* y = Pm + row * (long)C;
  int t = threadIdx.x;
  __shared__ float red[256];
  float v = (t < C) ? x[t] * scale : -1e30f;
  red[t] = v; __syncthreads();
  for (int k = 128; k > 0; k >>= 1) { if (t < k) red[t] = fmaxf(red[t], red[t + k]); __syncthreads(); }
  float mx = red[0]; __syncthreads();
  float e = (t < C) ? __expf(v - mx) : 0.f;
  red[t] = e; __syncthreads();
  for (int k = 128; k > 0; k >>= 1) { if (t < k) red[t] += red[t + k]; __syncthreads(); }
  float inv = 1.f / red[0];
  if (t < C) y[t] = (_Float16)(e * inv);
}

// out_row(f16) = LayerNorm(alpha * X + X2) over D
__global__ __launch_bounds__(256)
void k_res_ln(const _Float16* __restrict__ X, const _Float16* __restrict__ X2,
              const float* __restrict__ alpha, const float* __restrict__ g,
              const float* __restrict__ bb, _Float16* __restrict__ out, int D)
{
  long row = blockIdx.x;
  int t = threadIdx.x;
  const _Float16* xr  = X  + row * (long)D;
  const _Float16* x2r = X2 + row * (long)D;
  float a = alpha[0];
  __shared__ float red[256];
  float s = 0.f;
  for (int j = t; j < D; j += 256) s += a * (float)xr[j] + (float)x2r[j];
  red[t] = s; __syncthreads();
  for (int k = 128; k > 0; k >>= 1) { if (t < k) red[t] += red[t + k]; __syncthreads(); }
  float mean = red[0] / D; __syncthreads();
  float s2 = 0.f;
  for (int j = t; j < D; j += 256) {
    float y = a * (float)xr[j] + (float)x2r[j] - mean; s2 += y * y;
  }
  red[t] = s2; __syncthreads();
  for (int k = 128; k > 0; k >>= 1) { if (t < k) red[t] += red[t + k]; __syncthreads(); }
  float inv = rsqrtf(red[0] / D + 1e-5f);
  for (int j = t; j < D; j += 256) {
    float y = a * (float)xr[j] + (float)x2r[j];
    out[row * (long)D + j] = (_Float16)((y - mean) * inv * g[j] + bb[j]);
  }
}

__global__ void k_mean(const _Float16* __restrict__ in, float* __restrict__ out, int L, int J)
{
  int b = blockIdx.x, j = threadIdx.x;
  const _Float16* p = in + (long)b * L * J + j;
  float s = 0.f;
  for (int l = 0; l < L; ++l) s += (float)p[(long)l * J];
  out[(long)b * J + j] = s / L;
}

__global__ __launch_bounds__(256)
void k_add_ln(const float* __restrict__ Aa, const float* __restrict__ Bv,
              const float* __restrict__ g, const float* __restrict__ be,
              float* __restrict__ out, long ostride, int D)
{
  int b = blockIdx.x, t = threadIdx.x;
  __shared__ float red[256];
  float v = Aa[(long)b * D + t] + Bv[(long)b * D + t];
  red[t] = v; __syncthreads();
  for (int k = 128; k > 0; k >>= 1) { if (t < k) red[t] += red[t + k]; __syncthreads(); }
  float mean = red[0] / D; __syncthreads();
  float d = v - mean;
  red[t] = d * d; __syncthreads();
  for (int k = 128; k > 0; k >>= 1) { if (t < k) red[t] += red[t + k]; __syncthreads(); }
  float inv = rsqrtf(red[0] / D + 1e-5f);
  out[(long)b * ostride + t] = d * inv * g[t] + be[t];
}

// ---------------- host-side helpers -----------------------------------------
static inline void gemm(hipStream_t st,
    const _Float16* A, int lda, long sAb, long sAh,
    const _Float16* Bw, int ldb, long sBb, long sBh,
    const float* bias,
    const _Float16* Rres, int ldr, long sRb, long sRh,
    void* C, int ldc, long sCb, long sCh, int cfmt,
    int M, int N, int K, int batch, int H)
{
  int Nt = N / 32, NtG = (Nt + 7) / 8;
  dim3 grid((unsigned)((M / 32) * NtG), (unsigned)batch, 1);
  k_gemm<<<grid, 256, 0, st>>>(A, lda, sAb, sAh, Bw, ldb, sBb, sBh,
                               bias, Rres, ldr, sRb, sRh, C, ldc, sCb, sCh, cfmt,
                               M, N, K, H);
}

// PreFusionEnhancer for L=256 paths (txt / img_patch), D=768, H=4, dh=192
static void enhance_big(hipStream_t st, const float* xin, const float* pos,
  const _Float16* qW, const float* qB, const _Float16* kW, const float* kB,
  const _Float16* vW, const float* vB, const _Float16* oW, const float* oB,
  const float* alpha, const float* lng, const float* lnb,
  _Float16* OUT, _Float16* X, _Float16* Qb, _Float16* Kb, _Float16* Vb,
  float* S, _Float16* P16, _Float16* VT)
{
  const int ROWS = 16384;                       // B * L
  const long tot = (long)ROWS * 768;
  k_add_pos<<<(unsigned)((tot + 255) / 256), 256, 0, st>>>(xin, pos, X, tot, 256 * 768);
  gemm(st, X, 768, 0, 0, qW, 768, 0, 0, qB, nullptr, 0, 0, 0, Qb, 768, 0, 0, 1, ROWS, 768, 768, 1, 1);
  gemm(st, X, 768, 0, 0, kW, 768, 0, 0, kB, nullptr, 0, 0, 0, Kb, 768, 0, 0, 1, ROWS, 768, 768, 1, 1);
  gemm(st, X, 768, 0, 0, vW, 768, 0, 0, vB, nullptr, 0, 0, 0, Vb, 768, 0, 0, 1, ROWS, 768, 768, 1, 1);
  // S[b,h] = Q K^T : batched over 64*4, dh=192, f32 scores
  gemm(st, Qb, 768, 256L * 768, 192, Kb, 768, 256L * 768, 192, nullptr,
       nullptr, 0, 0, 0, S, 256, 4L * 65536, 65536, 0, 256, 256, 192, 256, 4);
  k_softmax<<<65536, 256, 0, st>>>(S, P16, 256, 0.07216878364870323f); // 1/sqrt(192)
  // VT (B,H,dh=192,L=256) <- V (B,L,768)
  k_transpose<<<dim3(8, 6, 256), 256, 0, st>>>(Vb, VT, 256, 768, 192, 4);
  // AO = P * V  (B of GEMM = VT rows, K-contiguous), head-concat into Qb
  gemm(st, P16, 256, 4L * 65536, 65536, VT, 256, 4L * 192 * 256, 192L * 256, nullptr,
       nullptr, 0, 0, 0, Qb, 768, 256L * 768, 192, 1, 256, 192, 256, 256, 4);
  // X2 = AO Wo^T + bo  (into Kb, free after scores)
  gemm(st, Qb, 768, 0, 0, oW, 768, 0, 0, oB, nullptr, 0, 0, 0, Kb, 768, 0, 0, 1, ROWS, 768, 768, 1, 1);
  k_res_ln<<<ROWS, 256, 0, st>>>(X, Kb, alpha, lng, lnb, OUT, 768);
}

// cross-modal MHA in joint space (J=256, H=4, dh=64)
static void cross_attn(hipStream_t st, const _Float16* Qs, const _Float16* Ks, const _Float16* Vs,
  const _Float16* qW, const float* qB, const _Float16* kW, const float* kB,
  const _Float16* vW, const float* vB, const _Float16* oW, const float* oB,
  _Float16* OUT, _Float16* Xs, float* S, _Float16* P16, _Float16* AO, _Float16* VT)
{
  const int ROWS = 16384;
  _Float16* qp = Xs;
  _Float16* kp = Xs + 16384L * 256;
  _Float16* vp = Xs + 2 * 16384L * 256;
  gemm(st, Qs, 256, 0, 0, qW, 256, 0, 0, qB, nullptr, 0, 0, 0, qp, 256, 0, 0, 1, ROWS, 256, 256, 1, 1);
  gemm(st, Ks, 256, 0, 0, kW, 256, 0, 0, kB, nullptr, 0, 0, 0, kp, 256, 0, 0, 1, ROWS, 256, 256, 1, 1);
  gemm(st, Vs, 256, 0, 0, vW, 256, 0, 0, vB, nullptr, 0, 0, 0, vp, 256, 0, 0, 1, ROWS, 256, 256, 1, 1);
  gemm(st, qp, 256, 256L * 256, 64, kp, 256, 256L * 256, 64, nullptr,
       nullptr, 0, 0, 0, S, 256, 4L * 65536, 65536, 0, 256, 256, 64, 256, 4);
  k_softmax<<<65536, 256, 0, st>>>(S, P16, 256, 0.125f); // 1/sqrt(64)
  k_transpose<<<dim3(8, 2, 256), 256, 0, st>>>(vp, VT, 256, 256, 64, 4);
  gemm(st, P16, 256, 4L * 65536, 65536, VT, 256, 4L * 64 * 256, 64L * 256, nullptr,
       nullptr, 0, 0, 0, AO, 256, 256L * 256, 64, 1, 256, 64, 256, 256, 4);
  gemm(st, AO, 256, 0, 0, oW, 256, 0, 0, oB, nullptr, 0, 0, 0, OUT, 256, 0, 0, 1, ROWS, 256, 256, 1, 1);
}

// ============================================================================
extern "C" void kernel_launch(void* const* d_in, const int* in_sizes, int n_in,
                              void* d_out, int out_size, void* d_ws, size_t ws_size,
                              hipStream_t stream)
{
  (void)in_sizes; (void)n_in; (void)out_size; (void)ws_size;
  const float* img_global = (const float*)d_in[IN_IMG_GLOBAL];
  const float* img_patch  = (const float*)d_in[IN_IMG_PATCH];
  const float* txt_feats  = (const float*)d_in[IN_TXT];
  auto P = [&](int i) { return (const float*)d_in[i]; };

  // ---- workspace carve-up: f32 first, then f16 arenas ----
  const long SZ_BIG = 16384L * 768;       // B*L x 768
  float* S   = (float*)d_ws;              // (B,H,256,256) f32 scores
  float* T0  = S + 64L * 4 * 256 * 256;   // (64,256) f32 x4
  float* T1  = T0 + 64L * 256;
  float* T2  = T1 + 64L * 256;
  float* T3  = T2 + 64L * 256;
  _Float16* hf = (_Float16*)(T3 + 64L * 256);
  _Float16* X    = hf;                    // pos-added input / cross q'k'v'
  _Float16* Qb   = X    + SZ_BIG;
  _Float16* Kb   = Qb   + SZ_BIG;
  _Float16* Vb   = Kb   + SZ_BIG;
  _Float16* TXT  = Vb   + SZ_BIG;
  _Float16* IPE  = TXT  + SZ_BIG;
  _Float16* P16  = IPE  + SZ_BIG;         // f16 softmax probabilities
  _Float16* VT   = P16  + 64L * 4 * 256 * 256;  // transposed V (B,H,dh,L)
  _Float16* IGE  = VT   + SZ_BIG;         // (64,768)
  _Float16* AT2I = IGE  + 64L * 768;      // (16384,256)
  _Float16* AI2T = AT2I + 16384L * 256;
  _Float16* AO2  = AI2T + 16384L * 256;   // cross-attn head-concat output
  _Float16* T4   = AO2  + 16384L * 256;   // (64,768) f16 x3 for ig path
  _Float16* T5   = T4 + 64L * 768;
  _Float16* T6   = T5 + 64L * 768;
  _Float16* W16  = T6 + 64L * 768;        // f16 weight arena (~7.8M halves)

  // ---- 0) one-shot weight conversion to f16 ----
  const long D2 = 768L * 768, JD = 256L * 768, J2 = 256L * 256;
  long woff = 0;
  auto CVT = [&](int idx, long n) -> const _Float16* {
    _Float16* dst = W16 + woff; woff += n;
    k_cvt<<<(unsigned)((n + 255) / 256), 256, 0, stream>>>(P(idx), dst, n);
    return dst;
  };
  const _Float16 *wTq = CVT(TXT_QW, D2), *wTk = CVT(TXT_KW, D2);
  const _Float16 *wTv = CVT(TXT_VW, D2), *wTo = CVT(TXT_OW, D2);
  const _Float16 *wPq = CVT(IP_QW, D2),  *wPk = CVT(IP_KW, D2);
  const _Float16 *wPv = CVT(IP_VW, D2),  *wPo = CVT(IP_OW, D2);
  const _Float16 *wGv = CVT(IG_VW, D2),  *wGo = CVT(IG_OW, D2);
  const _Float16 *wQT = CVT(QTXT_W, JD), *wKI = CVT(KIMG_W, JD), *wVI = CVT(VIMG_W, JD);
  const _Float16 *wQI = CVT(QIMG_W, JD), *wKT = CVT(KTXT_W, JD), *wVT2 = CVT(VTXT_W, JD);
  const _Float16 *wAq = CVT(T2I_QW, J2), *wAk = CVT(T2I_KW, J2);
  const _Float16 *wAv = CVT(T2I_VW, J2), *wAo = CVT(T2I_OW, J2);
  const _Float16 *wBq = CVT(I2T_QW, J2), *wBk = CVT(I2T_KW, J2);
  const _Float16 *wBv = CVT(I2T_VW, J2), *wBo = CVT(I2T_OW, J2);
  const _Float16 *wPP = CVT(PPROJ_W, JD), *wGP = CVT(GPROJ_W, JD), *wTP = CVT(TPROJ_W, JD);

  // ---- 1) enhance txt -> TXT ----
  enhance_big(stream, txt_feats, P(TXT_POS),
              wTq, P(TXT_QB), wTk, P(TXT_KB), wTv, P(TXT_VB), wTo, P(TXT_OB),
              P(TXT_ALPHA), P(TXT_LNG), P(TXT_LNB), TXT, X, Qb, Kb, Vb, S, P16, VT);

  // ---- 2) enhance img_patch -> IPE ----
  enhance_big(stream, img_patch, P(IP_POS),
              wPq, P(IP_QB), wPk, P(IP_KB), wPv, P(IP_VB), wPo, P(IP_OB),
              P(IP_ALPHA), P(IP_LNG), P(IP_LNB), IPE, X, Qb, Kb, Vb, S, P16, VT);

  // ---- 3) enhance img_global (L=1: softmax over 1 key => attn out == V) ----
  {
    long tot = 64L * 768;
    k_add_pos<<<(unsigned)((tot + 255) / 256), 256, 0, stream>>>(img_global, P(IG_POS), T4, tot, 768);
    gemm(stream, T4, 768, 0, 0, wGv, 768, 0, 0, P(IG_VB), nullptr, 0, 0, 0, T5, 768, 0, 0, 1, 64, 768, 768, 1, 1);
    gemm(stream, T5, 768, 0, 0, wGo, 768, 0, 0, P(IG_OB), nullptr, 0, 0, 0, T6, 768, 0, 0, 1, 64, 768, 768, 1, 1);
    k_res_ln<<<64, 256, 0, stream>>>(T4, T6, P(IG_ALPHA), P(IG_LNG), P(IG_LNB), IGE, 768);
  }

  // ---- 4) txt -> img cross attention (att_t2i) ----
  gemm(stream, TXT, 768, 0, 0, wQT, 768, 0, 0, P(QTXT_B), nullptr, 0, 0, 0, Qb, 256, 0, 0, 1, 16384, 256, 768, 1, 1);
  gemm(stream, IPE, 768, 0, 0, wKI, 768, 0, 0, P(KIMG_B), nullptr, 0, 0, 0, Kb, 256, 0, 0, 1, 16384, 256, 768, 1, 1);
  gemm(stream, IPE, 768, 0, 0, wVI, 768, 0, 0, P(VIMG_B), nullptr, 0, 0, 0, Vb, 256, 0, 0, 1, 16384, 256, 768, 1, 1);
  cross_attn(stream, Qb, Kb, Vb,
             wAq, P(T2I_QB), wAk, P(T2I_KB), wAv, P(T2I_VB), wAo, P(T2I_OB),
             AT2I, X, S, P16, AO2, VT);

  // ---- 5) img -> txt cross attention (att_i2t) ----
  gemm(stream, IPE, 768, 0, 0, wQI, 768, 0, 0, P(QIMG_B), nullptr, 0, 0, 0, Qb, 256, 0, 0, 1, 16384, 256, 768, 1, 1);
  gemm(stream, TXT, 768, 0, 0, wKT, 768, 0, 0, P(KTXT_B), nullptr, 0, 0, 0, Kb, 256, 0, 0, 1, 16384, 256, 768, 1, 1);
  gemm(stream, TXT, 768, 0, 0, wVT2, 768, 0, 0, P(VTXT_B), nullptr, 0, 0, 0, Vb, 256, 0, 0, 1, 16384, 256, 768, 1, 1);
  cross_attn(stream, Qb, Kb, Vb,
             wBq, P(I2T_QB), wBk, P(I2T_KB), wBv, P(I2T_VB), wBo, P(I2T_OB),
             AI2T, X, S, P16, AO2, VT);

  // ---- 6) final assembly into out (B, 258, 256) f32 ----
  float* out = (float*)d_out;
  // patches_fused -> out[:, 1:257, :]  (GEMM with fused att_i2t residual)
  gemm(stream, IPE, 768, 256L * 768, 0, wPP, 768, 0, 0, P(PPROJ_B),
       AI2T, 256, 256L * 256, 0, out + 256, 256, 258L * 256, 0, 0, 256, 256, 768, 64, 1);
  // x1 = LN(ig @ Wgp^T + b + mean_L(att_t2i)) -> out[:, 0, :]
  gemm(stream, IGE, 768, 0, 0, wGP, 768, 0, 0, P(GPROJ_B), nullptr, 0, 0, 0, T0, 256, 0, 0, 0, 64, 256, 768, 1, 1);
  k_mean<<<64, 256, 0, stream>>>(AT2I, T1, 256, 256);
  k_add_ln<<<64, 256, 0, stream>>>(T0, T1, P(LNIMG_G), P(LNIMG_B), out, 258L * 256, 256);
  // x2 = LN(txt[:,0] @ Wtp^T + b + mean_Np(att_i2t)) -> out[:, 257, :]
  gemm(stream, TXT, 256 * 768, 0, 0, wTP, 768, 0, 0, P(TPROJ_B), nullptr, 0, 0, 0, T2, 256, 0, 0, 0, 64, 256, 768, 1, 1);
  k_mean<<<64, 256, 0, stream>>>(AI2T, T3, 256, 256);
  k_add_ln<<<64, 256, 0, stream>>>(T2, T3, P(LNTXT_G), P(LNTXT_B), out + 257L * 256, 258L * 256, 256);
}
